// MultiLatentAttention_2138893714157
// MI455X (gfx1250) — matrix operand order repaired and checked by reading, not compiled, not run on hardware
//
#include <hip/hip_runtime.h>

// ---------------------------------------------------------------------------
// MLA (multi-latent attention) for gfx1250 / MI455X.
// All matmuls in bf16 via v_wmma_f32_16x16x32_bf16 (wave32 WMMA).
// GEMMs use a ping-pong double-buffered LDS pipeline + global_prefetch.
// ---------------------------------------------------------------------------

#define B_ 2
#define S_ 2048
#define D_ 2048
#define H_ 16
#define L_ 512
#define HD_ 128

typedef __attribute__((ext_vector_type(16))) __bf16 v16bf;
typedef __attribute__((ext_vector_type(8)))  float  v8f;

union ABf {
    v16bf v;
    unsigned int u[8];
};

__device__ __forceinline__ unsigned short f2bf(float f) {
    unsigned int u = __float_as_uint(f);
    u += 0x7FFFu + ((u >> 16) & 1u);           // round-to-nearest-even
    return (unsigned short)(u >> 16);
}

// --------------------------- conversion kernels ----------------------------

__global__ void cvt_f32_bf16_kernel(const float* __restrict__ in,
                                    unsigned short* __restrict__ out, size_t n) {
    for (size_t i = (size_t)blockIdx.x * blockDim.x + threadIdx.x; i < n;
         i += (size_t)gridDim.x * blockDim.x)
        out[i] = f2bf(in[i]);
}

// in [R][C] f32 -> out [C][R] bf16 (transpose; so GEMM B operand is [N][K])
__global__ void cvtT_f32_bf16_kernel(const float* __restrict__ in,
                                     unsigned short* __restrict__ out,
                                     int R, int C) {
    size_t n = (size_t)R * C;
    for (size_t i = (size_t)blockIdx.x * blockDim.x + threadIdx.x; i < n;
         i += (size_t)gridDim.x * blockDim.x) {
        int r = (int)(i / C);
        int c = (int)(i % C);
        out[(size_t)c * R + r] = f2bf(in[i]);
    }
}

// ------------------------------- layernorm ---------------------------------

__global__ __launch_bounds__(256) void ln_kernel(const float* __restrict__ in,
                                                 const float* __restrict__ g,
                                                 const float* __restrict__ bta,
                                                 unsigned short* __restrict__ out,
                                                 int L) {
    const int row  = blockIdx.x;
    const int tid  = threadIdx.x;
    const int wid  = tid >> 5;
    const int lane = tid & 31;
    const float* xr = in + (size_t)row * L;

    float s = 0.f, ss = 0.f;
    for (int i = tid; i < L; i += 256) {
        float v = xr[i];
        s += v; ss += v * v;
    }
    #pragma unroll
    for (int off = 16; off; off >>= 1) {
        s  += __shfl_xor(s,  off, 32);
        ss += __shfl_xor(ss, off, 32);
    }
    __shared__ float sh_s[8], sh_ss[8];
    if (lane == 0) { sh_s[wid] = s; sh_ss[wid] = ss; }
    __syncthreads();
    float ts = 0.f, tss = 0.f;
    #pragma unroll
    for (int w = 0; w < 8; ++w) { ts += sh_s[w]; tss += sh_ss[w]; }
    float mean = ts / (float)L;
    float var  = tss / (float)L - mean * mean;
    float inv  = rsqrtf(var + 1e-5f);
    for (int i = tid; i < L; i += 256)
        out[(size_t)row * L + i] = f2bf((xr[i] - mean) * inv * g[i] + bta[i]);
}

// ------------------------------ WMMA GEMM ----------------------------------
// C[M,N] = A[M,K](bf16,row-major) * B[K,N] + bias, with B given as Bt[N][K].
// Block: 128x128 tile, 8 waves in 2x4; wave: 64x32 -> 4x2 16x16 accumulators.
// Ping-pong LDS double buffering: global loads of tile t+1 overlap WMMA on t.

#define BM 128
#define BN 128
#define BK 32

__global__ __launch_bounds__(256) void gemm_bf16_kernel(
    const unsigned short* __restrict__ A,   // [M,K] bf16
    const unsigned short* __restrict__ Bt,  // [N,K] bf16
    const float* __restrict__ bias,         // [N]
    float* __restrict__ Cf,                 // used if !out_bf16
    unsigned short* __restrict__ Cb,        // used if  out_bf16
    int M, int N, int K, int out_bf16) {
    __shared__ __attribute__((aligned(16))) unsigned short lds_a[2][BM * BK];
    __shared__ __attribute__((aligned(16))) unsigned short lds_b[2][BN * BK];

    const int tid  = threadIdx.x;
    const int wid  = tid >> 5;
    const int lane = tid & 31;
    const int half = lane >> 4;
    const int ln16 = lane & 15;
    const int wm   = wid >> 2;               // 0..1
    const int wn   = wid & 3;                // 0..3
    const int tm   = blockIdx.x * BM;
    const int tn   = blockIdx.y * BN;

    v8f acc[4][2];
    #pragma unroll
    for (int i = 0; i < 4; ++i)
        #pragma unroll
        for (int j = 0; j < 2; ++j)
            #pragma unroll
            for (int r = 0; r < 8; ++r) acc[i][j][r] = 0.f;

    const unsigned int* Au = (const unsigned int*)A;
    const unsigned int* Bu = (const unsigned int*)Bt;

    // per-thread staging indices (8 uints each for A and B tiles)
    int srow[8], scol[8];
    #pragma unroll
    for (int i = 0; i < 8; ++i) {
        int idx = i * 256 + tid;               // 0..2047
        srow[i] = idx >> 4;                    // tile row
        scol[i] = idx & 15;                    // uint col (2 bf16 each)
    }

    const int T = K / BK;
    unsigned int ar[8], br[8];

    // preload tile 0
    #pragma unroll
    for (int i = 0; i < 8; ++i) {
        ar[i] = Au[((size_t)(tm + srow[i]) * K + 0) / 2 + scol[i]];
        br[i] = Bu[((size_t)(tn + srow[i]) * K + 0) / 2 + scol[i]];
    }
    {
        unsigned int* lau = (unsigned int*)lds_a[0];
        unsigned int* lbu = (unsigned int*)lds_b[0];
        #pragma unroll
        for (int i = 0; i < 8; ++i) {
            lau[i * 256 + tid] = ar[i];
            lbu[i * 256 + tid] = br[i];
        }
    }
    __syncthreads();

    for (int t = 0; t < T; ++t) {
        const int buf = t & 1;
        // issue global loads for tile t+1 (overlap with WMMA below)
        if (t + 1 < T) {
            int kb = (t + 1) * BK;
            #pragma unroll
            for (int i = 0; i < 8; ++i) {
                ar[i] = Au[((size_t)(tm + srow[i]) * K + kb) / 2 + scol[i]];
                br[i] = Bu[((size_t)(tn + srow[i]) * K + kb) / 2 + scol[i]];
            }
        }
        // prefetch tile t+2 into GL2 (global_prefetch_b8)
        if (t + 2 < T) {
            int kp = (t + 2) * BK;
            __builtin_prefetch(&Au[((size_t)(tm + srow[0]) * K + kp) / 2 + scol[0]], 0, 3);
            __builtin_prefetch(&Bu[((size_t)(tn + srow[0]) * K + kp) / 2 + scol[0]], 0, 3);
        }

        const unsigned int* lau = (const unsigned int*)lds_a[buf];
        const unsigned int* lbu = (const unsigned int*)lds_b[buf];
        ABf af[4];
        #pragma unroll
        for (int mi = 0; mi < 4; ++mi) {
            int m0 = wm * 64 + mi * 16 + ln16;
            #pragma unroll
            for (int v = 0; v < 8; ++v) {
                int k = ((v < 4) ? 2 * v : 8 + 2 * v) + 8 * half;   // A-frag layout
                af[mi].u[v] = lau[(m0 * BK + k) >> 1];
            }
        }
        #pragma unroll
        for (int ni = 0; ni < 2; ++ni) {
            ABf bfr;
            int n0 = wn * 32 + ni * 16 + ln16;
            #pragma unroll
            for (int v = 0; v < 8; ++v) {
                int k = ((v < 4) ? 2 * v : 8 + 2 * v) + 8 * half;   // B-frag layout
                bfr.u[v] = lbu[(n0 * BK + k) >> 1];
            }
            #pragma unroll
            for (int mi = 0; mi < 4; ++mi)
                acc[mi][ni] = __builtin_amdgcn_wmma_f32_16x16x32_bf16(
                    false, af[mi].v, false, bfr.v, (short)0, acc[mi][ni], false, false);
        }

        // store tile t+1 into the other buffer, then one barrier
        if (t + 1 < T) {
            unsigned int* wau = (unsigned int*)lds_a[buf ^ 1];
            unsigned int* wbu = (unsigned int*)lds_b[buf ^ 1];
            #pragma unroll
            for (int i = 0; i < 8; ++i) {
                wau[i * 256 + tid] = ar[i];
                wbu[i * 256 + tid] = br[i];
            }
        }
        __syncthreads();
    }

    #pragma unroll
    for (int mi = 0; mi < 4; ++mi)
        #pragma unroll
        for (int ni = 0; ni < 2; ++ni) {
            int col = tn + wn * 32 + ni * 16 + ln16;
            float bv = bias ? bias[col] : 0.f;
            #pragma unroll
            for (int r = 0; r < 8; ++r) {
                int row = tm + wm * 64 + mi * 16 + r + 8 * half;    // C/D layout
                float val = acc[mi][ni][r] + bv;
                size_t off = (size_t)row * N + col;
                if (out_bf16) Cb[off] = f2bf(val);
                else          Cf[off] = val;
            }
        }
}

// --------------------------- flash attention -------------------------------
// grid: (B*H, S/128); block 256 = 8 waves; wave owns 16 query rows.

__global__ __launch_bounds__(256) void mla_attn_kernel(
    const unsigned short* __restrict__ qb,   // [B*S, D] bf16
    const unsigned short* __restrict__ kvb,  // [B*S, 2D] bf16 (k | v)
    const unsigned char* __restrict__ mask,  // [B*S] (true = padded)
    unsigned short* __restrict__ ob) {       // [B*S, D] bf16
    const int KVW = 2 * D_;
    const float scale = 0.088388347648318447f;   // 1/sqrt(128)

    __shared__ __attribute__((aligned(16))) unsigned short lds_k[32 * HD_];
    __shared__ __attribute__((aligned(16))) unsigned short lds_v[32 * HD_];
    __shared__ __attribute__((aligned(16))) unsigned short lds_p[8 * 16 * 32];

    const int tid  = threadIdx.x;
    const int wid  = tid >> 5;
    const int lane = tid & 31;
    const int half = lane >> 4;
    const int ln16 = lane & 15;
    const int bh = blockIdx.x;
    const int b  = bh / H_;
    const int h  = bh % H_;
    const int qbase = blockIdx.y * 128;
    const int q0 = qbase + wid * 16;

    const unsigned int* qu  = (const unsigned int*)qb;
    const unsigned int* lku = (const unsigned int*)lds_k;
    const unsigned int* lpu = (const unsigned int*)lds_p;

    // Q fragments: 16 rows x 128 (4 x K=32)
    ABf qf[4];
    {
        size_t rowoff = ((size_t)(b * S_ + q0 + ln16)) * D_ + h * HD_;
        #pragma unroll
        for (int f = 0; f < 4; ++f)
            #pragma unroll
            for (int v = 0; v < 8; ++v) {
                int k = f * 32 + ((v < 4) ? 2 * v : 8 + 2 * v) + 8 * half;
                qf[f].u[v] = qu[(rowoff + k) >> 1];
            }
    }

    v8f o[8];
    #pragma unroll
    for (int i = 0; i < 8; ++i)
        #pragma unroll
        for (int r = 0; r < 8; ++r) o[i][r] = 0.f;
    float mrow[8], lrow[8];
    #pragma unroll
    for (int r = 0; r < 8; ++r) { mrow[r] = -1e30f; lrow[r] = 0.f; }

    const int kend = qbase + 128;
    for (int kb = 0; kb < kend; kb += 32) {
        // stage K / V tiles: 32 keys x 128 dims each (uint4 = 8 bf16)
        #pragma unroll
        for (int i = 0; i < 2; ++i) {
            int idx = i * 256 + tid;          // 0..511
            int r = idx >> 4;                 // key row 0..31
            int c = idx & 15;                 // 8-elem group
            size_t koff = ((size_t)(b * S_ + kb + r)) * KVW + h * HD_ + c * 8;
            ((uint4*)lds_k)[idx] = ((const uint4*)kvb)[koff >> 3];
            ((uint4*)lds_v)[idx] = ((const uint4*)kvb)[(koff + D_) >> 3];
            if (kb + 32 < kend) {             // prefetch next key block
                size_t pn = ((size_t)(b * S_ + kb + 32 + r)) * KVW + h * HD_ + c * 8;
                __builtin_prefetch(&((const uint4*)kvb)[pn >> 3], 0, 3);
            }
        }
        __syncthreads();

        if (kb <= q0 + 15) {
            // scores: 16 q x 32 k via 2 x (4 WMMAs)
            v8f sc[2];
            #pragma unroll
            for (int nf = 0; nf < 2; ++nf) {
                #pragma unroll
                for (int r = 0; r < 8; ++r) sc[nf][r] = 0.f;
                #pragma unroll
                for (int f = 0; f < 4; ++f) {
                    ABf kf;
                    #pragma unroll
                    for (int v = 0; v < 8; ++v) {
                        int k = f * 32 + ((v < 4) ? 2 * v : 8 + 2 * v) + 8 * half;
                        int n = nf * 16 + ln16;
                        kf.u[v] = lku[(n * HD_ + k) >> 1];
                    }
                    sc[nf] = __builtin_amdgcn_wmma_f32_16x16x32_bf16(
                        false, qf[f].v, false, kf.v, (short)0, sc[nf], false, false);
                }
            }
            // mask + scale
            float sv[2][8];
            #pragma unroll
            for (int nf = 0; nf < 2; ++nf)
                #pragma unroll
                for (int r = 0; r < 8; ++r) {
                    int key = kb + nf * 16 + ln16;
                    int qr  = q0 + r + 8 * half;
                    float x = sc[nf][r] * scale;
                    if (key > qr || mask[b * S_ + key]) x = -1e30f;
                    sv[nf][r] = x;
                }
            // online softmax per row, P written to per-wave LDS scratch
            #pragma unroll
            for (int r = 0; r < 8; ++r) {
                float t = fmaxf(sv[0][r], sv[1][r]);
                #pragma unroll
                for (int off = 1; off <= 8; off <<= 1)
                    t = fmaxf(t, __shfl_xor(t, off, 32));
                float mn   = fmaxf(mrow[r], t);
                float corr = __expf(mrow[r] - mn);
                float p0 = (sv[0][r] < -5e29f) ? 0.f : __expf(sv[0][r] - mn);
                float p1 = (sv[1][r] < -5e29f) ? 0.f : __expf(sv[1][r] - mn);
                float ps = p0 + p1;
                #pragma unroll
                for (int off = 1; off <= 8; off <<= 1)
                    ps += __shfl_xor(ps, off, 32);
                lrow[r] = lrow[r] * corr + ps;
                mrow[r] = mn;
                #pragma unroll
                for (int hf = 0; hf < 8; ++hf) o[hf][r] *= corr;
                int base = wid * 512 + (r + 8 * half) * 32;
                lds_p[base + ln16]      = f2bf(p0);
                lds_p[base + 16 + ln16] = f2bf(p1);
            }
            asm volatile("s_wait_dscnt 0" ::: "memory");
            // reload P in A-fragment layout
            ABf pf;
            #pragma unroll
            for (int v = 0; v < 8; ++v) {
                int k = ((v < 4) ? 2 * v : 8 + 2 * v) + 8 * half;
                pf.u[v] = lpu[(wid * 512 + ln16 * 32 + k) >> 1];
            }
            // O += P * V
            #pragma unroll
            for (int hf = 0; hf < 8; ++hf) {
                ABf vf;
                #pragma unroll
                for (int v = 0; v < 8; ++v) {
                    int kl = ((v < 4) ? 2 * v : 8 + 2 * v) + 8 * half;
                    int n  = hf * 16 + ln16;
                    unsigned int lo = lds_v[kl * HD_ + n];
                    unsigned int hi = lds_v[(kl + 1) * HD_ + n];
                    vf.u[v] = lo | (hi << 16);
                }
                o[hf] = __builtin_amdgcn_wmma_f32_16x16x32_bf16(
                    false, pf.v, false, vf.v, (short)0, o[hf], false, false);
            }
        }
        __syncthreads();
    }

    float invl[8];
    #pragma unroll
    for (int r = 0; r < 8; ++r) invl[r] = (lrow[r] > 0.f) ? 1.f / lrow[r] : 0.f;
    #pragma unroll
    for (int hf = 0; hf < 8; ++hf)
        #pragma unroll
        for (int r = 0; r < 8; ++r) {
            size_t off = ((size_t)(b * S_ + q0 + r + 8 * half)) * D_ +
                         h * HD_ + hf * 16 + ln16;
            ob[off] = f2bf(o[hf][r] * invl[r]);
        }
}

// ------------------------------- launcher ----------------------------------

extern "C" void kernel_launch(void* const* d_in, const int* in_sizes, int n_in,
                              void* d_out, int out_size, void* d_ws, size_t ws_size,
                              hipStream_t stream) {
    (void)in_sizes; (void)n_in; (void)out_size; (void)ws_size;
    const float* x        = (const float*)d_in[0];
    const unsigned char* mask = (const unsigned char*)d_in[1];
    const float* wq_down  = (const float*)d_in[2];
    const float* bq_down  = (const float*)d_in[3];
    const float* gq_ln    = (const float*)d_in[4];
    const float* bq_ln    = (const float*)d_in[5];
    const float* wq_up    = (const float*)d_in[6];
    const float* bq_up    = (const float*)d_in[7];
    const float* wkv_down = (const float*)d_in[8];
    const float* bkv_down = (const float*)d_in[9];
    const float* gkv_ln   = (const float*)d_in[10];
    const float* bkv_ln   = (const float*)d_in[11];
    const float* wkv_up   = (const float*)d_in[12];
    const float* bkv_up   = (const float*)d_in[13];
    const float* w_out    = (const float*)d_in[14];
    const float* b_out    = (const float*)d_in[15];

    const size_t BS = (size_t)B_ * S_;   // 4096
    char* w = (char*)d_ws;
    size_t off = 0;
    auto take = [&](size_t bytes) {
        char* p = w + off;
        off += (bytes + 255) & ~(size_t)255;
        return p;
    };
    unsigned short* xb     = (unsigned short*)take(BS * D_ * 2);
    unsigned short* wqdT   = (unsigned short*)take((size_t)L_ * D_ * 2);
    unsigned short* wquT   = (unsigned short*)take((size_t)L_ * D_ * 2);
    unsigned short* wkvdT  = (unsigned short*)take((size_t)L_ * D_ * 2);
    unsigned short* wkvuT  = (unsigned short*)take((size_t)L_ * 2 * D_ * 2);
    unsigned short* woT    = (unsigned short*)take((size_t)D_ * D_ * 2);
    float*          latraw = (float*)take(BS * L_ * 4);
    unsigned short* latbf  = (unsigned short*)take(BS * L_ * 2);
    unsigned short* qbuf   = (unsigned short*)take(BS * D_ * 2);
    unsigned short* kvbuf  = (unsigned short*)take(BS * 2 * D_ * 2);
    unsigned short* aob    = (unsigned short*)take(BS * D_ * 2);

    // conversions
    cvt_f32_bf16_kernel<<<2048, 256, 0, stream>>>(x, xb, BS * D_);
    cvtT_f32_bf16_kernel<<<2048, 256, 0, stream>>>(wq_down,  wqdT,  D_, L_);
    cvtT_f32_bf16_kernel<<<2048, 256, 0, stream>>>(wq_up,    wquT,  L_, D_);
    cvtT_f32_bf16_kernel<<<2048, 256, 0, stream>>>(wkv_down, wkvdT, D_, L_);
    cvtT_f32_bf16_kernel<<<2048, 256, 0, stream>>>(wkv_up,   wkvuT, L_, 2 * D_);
    cvtT_f32_bf16_kernel<<<2048, 256, 0, stream>>>(w_out,    woT,   D_, D_);

    const int M = (int)BS;
    // q path
    gemm_bf16_kernel<<<dim3(M / BM, L_ / BN), 256, 0, stream>>>(
        xb, wqdT, bq_down, latraw, nullptr, M, L_, D_, 0);
    ln_kernel<<<M, 256, 0, stream>>>(latraw, gq_ln, bq_ln, latbf, L_);
    gemm_bf16_kernel<<<dim3(M / BM, D_ / BN), 256, 0, stream>>>(
        latbf, wquT, bq_up, nullptr, qbuf, M, D_, L_, 1);
    // kv path
    gemm_bf16_kernel<<<dim3(M / BM, L_ / BN), 256, 0, stream>>>(
        xb, wkvdT, bkv_down, latraw, nullptr, M, L_, D_, 0);
    ln_kernel<<<M, 256, 0, stream>>>(latraw, gkv_ln, bkv_ln, latbf, L_);
    gemm_bf16_kernel<<<dim3(M / BM, (2 * D_) / BN), 256, 0, stream>>>(
        latbf, wkvuT, bkv_up, nullptr, kvbuf, M, 2 * D_, L_, 1);
    // attention
    mla_attn_kernel<<<dim3(B_ * H_, S_ / 128), 256, 0, stream>>>(
        qbuf, kvbuf, mask, aob);
    // output projection
    gemm_bf16_kernel<<<dim3(M / BM, D_ / BN), 256, 0, stream>>>(
        aob, woT, b_out, (float*)d_out, nullptr, M, D_, D_, 0);
}
